// DiffMultiHeadedAttention_51522427683068
// MI455X (gfx1250) — matrix operand order repaired and checked by reading, not compiled, not run on hardware
//
#include <hip/hip_runtime.h>
#include <hip/hip_bf16.h>
#include <math.h>
#include <stdint.h>

// ---------------------------------------------------------------------------
// Types for CDNA5 WMMA (wave32)
// ---------------------------------------------------------------------------
typedef __attribute__((ext_vector_type(8)))  float  v8f;
typedef __attribute__((ext_vector_type(16))) __bf16 v16bf;

union BF16Frag { v16bf v; unsigned short u[16]; };

#define HIDDEN 1024
#define NHEAD  16      // H
#define HEADD  64
#define HALFD  32
#define BB     4
#define TT     1024
#define NNK    1024

__device__ __forceinline__ unsigned short f2bf(float f) {
  unsigned int u = __float_as_uint(f);
  unsigned int r = (u + 0x7FFFu + ((u >> 16) & 1u)) >> 16;   // RNE
  return (unsigned short)r;
}

__device__ __forceinline__ v8f wmma_bf16(v16bf a, v16bf b, v8f c) {
  return __builtin_amdgcn_wmma_f32_16x16x32_bf16(false, a, false, b,
                                                 (short)0, c, false, false);
}

// Async global->LDS copy, 16 bytes per lane (GVS mode), tracked by ASYNCcnt.
__device__ __forceinline__ void async_b128(unsigned lds_byte_off,
                                           const void* gbase,
                                           unsigned gbyte_off) {
  asm volatile("global_load_async_to_lds_b128 %0, %1, %2"
               :
               : "v"(lds_byte_off), "v"(gbyte_off),
                 "s"((unsigned long long)(uintptr_t)gbase)
               : "memory");
}

__device__ __forceinline__ void wait_async0() {
  asm volatile("s_wait_asynccnt 0x0" ::: "memory");
}

__device__ __forceinline__ unsigned lds_off32(const void* p) {
  return (unsigned)(uintptr_t)p;   // low 32 bits of generic addr = LDS offset
}

// ---------------------------------------------------------------------------
// Kernel 0: lambda_full = exp(sum(lq1*lk1)) - exp(sum(lq2*lk2)) + 0.8
// ---------------------------------------------------------------------------
__global__ void lambda_kernel(const float* __restrict__ lq1,
                              const float* __restrict__ lq2,
                              const float* __restrict__ lk1,
                              const float* __restrict__ lk2,
                              float* __restrict__ lam) {
  int l = threadIdx.x;                 // 32 lanes == HALFD
  float p1 = lq1[l] * lk1[l];
  float p2 = lq2[l] * lk2[l];
#pragma unroll
  for (int off = 16; off; off >>= 1) {
    p1 += __shfl_xor(p1, off, 32);
    p2 += __shfl_xor(p2, off, 32);
  }
  if (l == 0) *lam = __expf(p1) - __expf(p2) + 0.8f;
}

// ---------------------------------------------------------------------------
// Kernel 1: one-shot f32 -> bf16 conversion (vectorized, RNE)
// ---------------------------------------------------------------------------
__global__ __launch_bounds__(256)
void cvt_bf16(const float* __restrict__ in, unsigned short* __restrict__ out,
              int n4) {
  int i = blockIdx.x * 256 + threadIdx.x;          // one float4 each
  if (i < n4) {
    float4 f = reinterpret_cast<const float4*>(in)[i];
    uint2 p;
    p.x = (unsigned)f2bf(f.x) | ((unsigned)f2bf(f.y) << 16);
    p.y = (unsigned)f2bf(f.z) | ((unsigned)f2bf(f.w) << 16);
    reinterpret_cast<uint2*>(out)[i] = p;
  }
}

// ---------------------------------------------------------------------------
// Kernel 2: C(bf16, MxN) = A(bf16, MxK) @ W^T(bf16 stored NxK) + bias(f32)
// 256 thr (8 waves), block tile 128x256, K-step 32, double-buffered async
// staging.  Wave grid 2x4: each wave computes a 64x64 tile = 4x4 WMMA tiles.
// ---------------------------------------------------------------------------
__global__ __launch_bounds__(256)
void proj_gemm(const unsigned short* __restrict__ A,
               const unsigned short* __restrict__ W,
               const float* __restrict__ bias,
               unsigned short* __restrict__ C,
               int M, int N, int K) {
  __shared__ __align__(16) unsigned short As[2][128][32];   // 16 KB
  __shared__ __align__(16) unsigned short Ws[2][256][32];   // 32 KB

  const int tid  = threadIdx.x;
  const int lane = tid & 31;
  const int wave = tid >> 5;
  const int wm   = wave & 1;            // 0..1 -> 64-row strip
  const int wn   = wave >> 1;           // 0..3 -> 64-col strip
  const int bm   = blockIdx.x * 128;
  const int bn   = blockIdx.y * 256;
  const int l16  = lane & 15;
  const int hi   = (lane >> 4) & 1;
  const int hi8  = hi * 8;
  const int hi16 = hi * 16;

  const unsigned asb = lds_off32(&As[0][0][0]);
  const unsigned wsb = lds_off32(&Ws[0][0][0]);

  // Stage one A(128x32) + W(256x32) bf16 tile pair into buffer `buf`.
  // 512 + 1024 16-byte chunks; each of 256 threads issues 2 A + 4 W copies.
  auto stage = [&](int buf, int k0) {
#pragma unroll
    for (int i = 0; i < 2; ++i) {
      int ck = tid + i * 256;              // 0..511
      int r = ck >> 2, c = ck & 3;
      unsigned ga = (unsigned)(((bm + r) * K + k0 + c * 8) * 2);
      async_b128(asb + (unsigned)buf * 8192u + (unsigned)ck * 16u, A, ga);
    }
#pragma unroll
    for (int i = 0; i < 4; ++i) {
      int ck = tid + i * 256;              // 0..1023
      int r = ck >> 2, c = ck & 3;
      unsigned gw = (unsigned)(((bn + r) * K + k0 + c * 8) * 2);
      async_b128(wsb + (unsigned)buf * 16384u + (unsigned)ck * 16u, W, gw);
    }
  };

  v8f acc[4][4];
#pragma unroll
  for (int mi = 0; mi < 4; ++mi)
#pragma unroll
    for (int ni = 0; ni < 4; ++ni)
#pragma unroll
      for (int r = 0; r < 8; ++r) acc[mi][ni][r] = 0.0f;

  stage(0, 0);
  int cur = 0;
  for (int k0 = 0; k0 < K; k0 += 32) {
    wait_async0();          // my async copies into buffer `cur` are done
    __syncthreads();        // everyone's are done; prev compute buf is free
    if (k0 + 32 < K) stage(cur ^ 1, k0 + 32);

    BF16Frag bfr[4];
#pragma unroll
    for (int ni = 0; ni < 4; ++ni) {
      int n = wn * 64 + ni * 16 + l16;
#pragma unroll
      for (int j = 0; j < 16; ++j) bfr[ni].u[j] = Ws[cur][n][((j < 8) ? j : j + 8) + hi8];
    }
    BF16Frag afr[4];
#pragma unroll
    for (int mi = 0; mi < 4; ++mi) {
      int m = wm * 64 + mi * 16 + l16;
#pragma unroll
      for (int j = 0; j < 16; ++j) {
        int kk = ((j < 8) ? j : j + 8) + hi8;
        afr[mi].u[j] = As[cur][m][kk];
      }
    }
#pragma unroll
    for (int mi = 0; mi < 4; ++mi)
#pragma unroll
      for (int ni = 0; ni < 4; ++ni)
        acc[mi][ni] = wmma_bf16(afr[mi].v, bfr[ni].v, acc[mi][ni]);
    cur ^= 1;
  }

#pragma unroll
  for (int mi = 0; mi < 4; ++mi)
#pragma unroll
    for (int ni = 0; ni < 4; ++ni) {
      int col = bn + wn * 64 + ni * 16 + l16;
      float b = bias[col];
#pragma unroll
      for (int r = 0; r < 8; ++r) {
        int row = bm + wm * 64 + mi * 16 + r + hi8;
        C[(size_t)row * N + col] = f2bf(acc[mi][ni][r] + b);
      }
    }
}

// ---------------------------------------------------------------------------
// Kernel 3: fused differential flash-attention.
// Grid: (T/128, H, B).  Block: 256 thr (8 waves).  Each wave: 16 query rows,
// both sub-heads (2h, 2h+1) with independent online softmax; K/V staged via
// double-buffered async-to-LDS in 64-key steps; out = O0/l0 - lambda * O1/l1.
// ---------------------------------------------------------------------------
__global__ __launch_bounds__(256)
void diff_attn(const unsigned short* __restrict__ qb,   // [B*T*1024] bf16
               const unsigned short* __restrict__ kb,   // [B*N*1024] bf16
               const unsigned short* __restrict__ vb,   // [B*N*1024] bf16
               const float* __restrict__ lam,
               float* __restrict__ out) {               // [B*T*1024] f32
  __shared__ __align__(16) unsigned short Ks[2][2][64][32]; // 16 KB
  __shared__ __align__(16) unsigned short Vs[2][64][64];    // 16 KB
  __shared__ __align__(16) unsigned short Ps[8][16][64];    // 16 KB

  const int tid  = threadIdx.x;
  const int lane = tid & 31;
  const int wave = tid >> 5;
  const int b    = blockIdx.z;
  const int h    = blockIdx.y;
  const int t0   = blockIdx.x * 128;
  const int tw   = t0 + wave * 16;
  const int l16  = lane & 15;
  const int hi   = (lane >> 4) & 1;
  const int hi8  = hi * 8;
  const int hi16 = hi * 16;
  const float sc = 0.17677669529663687f;  // 1/sqrt(32)

  const unsigned ksb = lds_off32(&Ks[0][0][0][0]);
  const unsigned vsb = lds_off32(&Vs[0][0][0]);

  // Stage K (2 sub-heads x 64x32) + V (64x64) for key block n0 into `buf`.
  auto stage = [&](int buf, int n0) {
#pragma unroll
    for (int i = 0; i < 2; ++i) {
      int ck = tid + i * 256;                      // 0..511
      {
        int s = ck >> 8, rem = ck & 255;
        int r = rem >> 2, c = rem & 3;
        unsigned g = (unsigned)(((b * NNK + n0 + r) * HIDDEN
                      + (2 * h + s) * HALFD + c * 8) * 2);
        async_b128(ksb + (unsigned)buf * 8192u + (unsigned)ck * 16u, kb, g);
      }
      {
        int r = ck >> 3, c = ck & 7;
        unsigned g = (unsigned)(((b * NNK + n0 + r) * HIDDEN
                      + h * HEADD + c * 8) * 2);
        async_b128(vsb + (unsigned)buf * 8192u + (unsigned)ck * 16u, vb, g);
      }
    }
  };

  // Q fragments for both sub-heads (A-matrix 16x32 layout), loaded once.
  BF16Frag qf[2];
#pragma unroll
  for (int s = 0; s < 2; ++s) {
    const unsigned short* qp =
        qb + (size_t)(b * TT + tw + l16) * HIDDEN + (2 * h + s) * HALFD;
#pragma unroll
    for (int j = 0; j < 16; ++j) {
      int d = ((j < 8) ? j : j + 8) + hi8;
      qf[s].u[j] = qp[d];
    }
  }

  v8f O[2][4];
  float mrun[2][8], lrun[2][8];
#pragma unroll
  for (int s = 0; s < 2; ++s) {
#pragma unroll
    for (int c = 0; c < 4; ++c)
#pragma unroll
      for (int r = 0; r < 8; ++r) O[s][c][r] = 0.0f;
#pragma unroll
    for (int r = 0; r < 8; ++r) { mrun[s][r] = -1e30f; lrun[s][r] = 0.0f; }
  }

  const int nb_end = (t0 + 128) / 64;   // causal horizon for this block
  stage(0, 0);
  int cur = 0;
  for (int nb = 0; nb < nb_end; ++nb) {
    const int n0 = nb * 64;
    wait_async0();
    __syncthreads();
    if (nb + 1 < nb_end) stage(cur ^ 1, (nb + 1) * 64);

    if (n0 <= tw + 15) {                 // wave-uniform: EXEC all-ones inside
      const bool need_mask = (n0 + 63 > tw);   // wave-uniform diagonal check
#pragma unroll
      for (int s = 0; s < 2; ++s) {
        // S = Q K^T for four 16-key tiles
        v8f S[4];
#pragma unroll
        for (int j = 0; j < 4; ++j) {
          BF16Frag kf;
#pragma unroll
          for (int jj = 0; jj < 16; ++jj)
            kf.u[jj] = Ks[cur][s][j * 16 + l16][((jj < 8) ? jj : jj + 8) + hi8];
          v8f z;
#pragma unroll
          for (int r = 0; r < 8; ++r) z[r] = 0.0f;
          S[j] = wmma_bf16(qf[s].v, kf.v, z);
        }
        // scale (+ causal mask only on diagonal blocks) + per-row max
        float tmax[8];
#pragma unroll
        for (int r = 0; r < 8; ++r) tmax[r] = -1e30f;
        if (need_mask) {
#pragma unroll
          for (int j = 0; j < 4; ++j)
#pragma unroll
            for (int r = 0; r < 8; ++r) {
              int row = tw + r + hi8;
              int col = n0 + j * 16 + l16;
              float e = S[j][r] * sc;
              if (col > row) e = -1e30f;
              S[j][r] = e;
              tmax[r] = fmaxf(tmax[r], e);
            }
        } else {
#pragma unroll
          for (int j = 0; j < 4; ++j)
#pragma unroll
            for (int r = 0; r < 8; ++r) {
              float e = S[j][r] * sc;
              S[j][r] = e;
              tmax[r] = fmaxf(tmax[r], e);
            }
        }
#pragma unroll
        for (int off = 1; off < 16; off <<= 1)
#pragma unroll
          for (int r = 0; r < 8; ++r)
            tmax[r] = fmaxf(tmax[r], __shfl_xor(tmax[r], off, 16));
        // online-softmax rescale
#pragma unroll
        for (int r = 0; r < 8; ++r) {
          float mnew = fmaxf(mrun[s][r], tmax[r]);
          float corr = __expf(mrun[s][r] - mnew);
          mrun[s][r] = mnew;
          lrun[s][r] *= corr;
#pragma unroll
          for (int c = 0; c < 4; ++c) O[s][c][r] *= corr;
        }
        // P = exp(S - m), row sums
        float ps[8];
#pragma unroll
        for (int r = 0; r < 8; ++r) ps[r] = 0.0f;
#pragma unroll
        for (int j = 0; j < 4; ++j)
#pragma unroll
          for (int r = 0; r < 8; ++r) {
            float p = __expf(S[j][r] - mrun[s][r]);
            S[j][r] = p;
            ps[r] += p;
          }
#pragma unroll
        for (int off = 1; off < 16; off <<= 1)
#pragma unroll
          for (int r = 0; r < 8; ++r) ps[r] += __shfl_xor(ps[r], off, 16);
#pragma unroll
        for (int r = 0; r < 8; ++r) lrun[s][r] += ps[r];

        // C-layout -> A-layout transpose of P through per-wave LDS scratch
#pragma unroll
        for (int j = 0; j < 4; ++j)
#pragma unroll
          for (int r = 0; r < 8; ++r)
            Ps[wave][r + hi8][j * 16 + l16] = f2bf(S[j][r]);
        asm volatile("s_wait_dscnt 0x0" ::: "memory");

        // O += P(16x64) @ V(64x64): 2 K-steps x 4 output tiles
#pragma unroll
        for (int k2 = 0; k2 < 2; ++k2) {
          BF16Frag pf;
#pragma unroll
          for (int jj = 0; jj < 16; ++jj) {
            int kk = k2 * 32 + ((jj < 8) ? jj : jj + 8) + hi8;
            pf.u[jj] = Ps[wave][l16][kk];
          }
#pragma unroll
          for (int c = 0; c < 4; ++c) {
            BF16Frag vf;
#pragma unroll
            for (int jj = 0; jj < 16; ++jj)
              vf.u[jj] = Vs[cur][k2 * 32 + ((jj < 8) ? jj : jj + 8) + hi8][c * 16 + l16];
            O[s][c] = wmma_bf16(pf.v, vf.v, O[s][c]);
          }
        }
      }
    }
    cur ^= 1;
  }

  // Epilogue: out = O0/l0 - lambda * O1/l1
  const float lamv = *lam;
#pragma unroll
  for (int c = 0; c < 4; ++c)
#pragma unroll
    for (int r = 0; r < 8; ++r) {
      int row = tw + r + hi8;
      int col = h * HEADD + c * 16 + l16;
      float o0 = O[0][c][r] / lrun[0][r];
      float o1 = O[1][c][r] / lrun[1][r];
      out[(size_t)(b * TT + row) * HIDDEN + col] = o0 - lamv * o1;
    }
}

// ---------------------------------------------------------------------------
// Host launch
// ---------------------------------------------------------------------------
extern "C" void kernel_launch(void* const* d_in, const int* in_sizes, int n_in,
                              void* d_out, int out_size, void* d_ws, size_t ws_size,
                              hipStream_t stream) {
  const float* enc = (const float*)d_in[0];
  const float* x   = (const float*)d_in[1];
  const float* Wq  = (const float*)d_in[2];
  const float* bq  = (const float*)d_in[3];
  const float* Wk  = (const float*)d_in[4];
  const float* bk  = (const float*)d_in[5];
  const float* Wv  = (const float*)d_in[6];
  const float* bv  = (const float*)d_in[7];
  const float* lq1 = (const float*)d_in[8];
  const float* lq2 = (const float*)d_in[9];
  const float* lk1 = (const float*)d_in[10];
  const float* lk2 = (const float*)d_in[11];
  float* out = (float*)d_out;

  const size_t act = (size_t)BB * TT * HIDDEN;       // 4M elems
  const size_t wsz = (size_t)HIDDEN * HIDDEN;        // 1M elems
  unsigned short* qws = (unsigned short*)d_ws;       // bf16 q/k/v
  unsigned short* kws = qws + act;
  unsigned short* vws = kws + act;
  unsigned short* xb  = vws + act;                   // bf16 copies of inputs
  unsigned short* eb  = xb + act;
  unsigned short* wqb = eb + act;
  unsigned short* wkb = wqb + wsz;
  unsigned short* wvb = wkb + wsz;
  float* lam = (float*)(wvb + wsz);

  lambda_kernel<<<1, 32, 0, stream>>>(lq1, lq2, lk1, lk2, lam);

  // one-shot f32 -> bf16 conversions
  cvt_bf16<<<(int)(act / 4 / 256), 256, 0, stream>>>(x,   xb, (int)(act / 4));
  cvt_bf16<<<(int)(act / 4 / 256), 256, 0, stream>>>(enc, eb, (int)(act / 4));
  cvt_bf16<<<(int)(wsz / 4 / 256), 256, 0, stream>>>(Wq,  wqb, (int)(wsz / 4));
  cvt_bf16<<<(int)(wsz / 4 / 256), 256, 0, stream>>>(Wk,  wkb, (int)(wsz / 4));
  cvt_bf16<<<(int)(wsz / 4 / 256), 256, 0, stream>>>(Wv,  wvb, (int)(wsz / 4));

  dim3 gg(32, 4);   // M=4096/128, N=1024/256
  proj_gemm<<<gg, 256, 0, stream>>>(xb, wqb, bq, qws, BB * TT,  HIDDEN, HIDDEN);
  proj_gemm<<<gg, 256, 0, stream>>>(eb, wkb, bk, kws, BB * NNK, HIDDEN, HIDDEN);
  proj_gemm<<<gg, 256, 0, stream>>>(eb, wvb, bv, vws, BB * NNK, HIDDEN, HIDDEN);

  diff_attn<<<dim3(TT / 128, NHEAD, BB), 256, 0, stream>>>(qws, kws, vws, lam, out);
}
